// FCA_59571196395682
// MI455X (gfx1250) — matrix operand order repaired and checked by reading, not compiled, not run on hardware
//
#include <hip/hip_runtime.h>

// ---------------------------------------------------------------------------
// FCA pipeline for MI455X (gfx1250, wave32, WMMA + TDM).
//   ca  = Dc   @ ip          (channel DCT-II, 64x64, fp32 WMMA 16x16x4)
//   ctk = top8_c(ca); gate = sigmoid(conv3x3_8->1(ctk)+b)
//   x   = Minv @ (ca*gate)   (channel DCT-III inverse, fp32 WMMA; also emits
//                             an f16 copy of x for the TDM-staged spatial DCT)
//   per (b,c): dct2 = D256 @ (x_img @ D256^T) fused in LDS (f16 WMMA 16x16x32)
//              image staged into LDS by tensor_load_to_lds (HW row padding)
//              topc = top8(dct2)            (never materialized to HBM)
//   w = sigmoid(relu(topc@fc1^T)@fc2^T);  out = x * w
// ---------------------------------------------------------------------------

typedef __attribute__((ext_vector_type(16))) _Float16 v16h;
typedef __attribute__((ext_vector_type(8)))  _Float16 v8h;
typedef __attribute__((ext_vector_type(8)))  float    v8f;
typedef __attribute__((ext_vector_type(4)))  float    v4f;
typedef __attribute__((ext_vector_type(2)))  float    v2f;
typedef __attribute__((ext_vector_type(4)))  unsigned int u32x4;
typedef __attribute__((ext_vector_type(8)))  int      i32x8;
typedef __attribute__((ext_vector_type(4)))  int      i32x4;

union V16 { v16h v; v8h h[2]; };

#if __has_include(<hip/amd_detail/amd_gfx1250_TDM.h>)
#define TDM_6ARG 1          // amdgpu-toolchain (clang-23 / therock headers)
#else
#define TDM_6ARG 0          // ROCm 7.2 (clang-22): 5-arg builtin
#endif
#define HAVE_TDM (__has_builtin(__builtin_amdgcn_tensor_load_to_lds) && \
                  __has_builtin(__builtin_amdgcn_s_wait_tensorcnt))

#define B_   4
#define C_   64
#define H_   256
#define W_   256
#define HW   65536            // H_*W_
#define CHW  (C_*HW)
#define TOT  (B_*CHW)
#define LDX  264              // f16 row pitch in LDS (256 + 8 pad -> bank skew)
#define PI_F 3.14159265358979323846f

__device__ __forceinline__ void top8_insert(float (&t)[8], float v) {
  if (v > t[7]) {
    t[7] = v;
    #pragma unroll
    for (int i = 7; i > 0; --i) {          // bubble the new value up
      float a = t[i-1], b = t[i];
      t[i-1] = fmaxf(a, b);
      t[i]   = fminf(a, b);
    }
  }
}

// ---- constant-matrix generation (exact integer phase reduction) -----------
__global__ void gen_mats(float* __restrict__ Dc, float* __restrict__ Minv,
                         _Float16* __restrict__ Drow) {
  int t = blockIdx.x * 256 + threadIdx.x;           // 65536 threads
  if (t < 64 * 64) {
    int k = t >> 6, c = t & 63;
    int m = (k * (2 * c + 1)) & 255;                // cos(pi*m/128), period 256
    Dc[t] = (1.0f / 64.0f) * __cosf((float)m * (PI_F / 128.0f));
    int n = t >> 6, kk = t & 63;                    // Minv[n][kk] (DCT-III)
    int m2 = (kk * (2 * n + 1)) & 255;
    Minv[t] = (kk == 0) ? 1.0f : 2.0f * __cosf((float)m2 * (PI_F / 128.0f));
  }
  {
    int k = t >> 8, n = t & 255;
    int m = (k * (2 * n + 1)) & 1023;               // cos(pi*m/512), period 1024
    Drow[t] = (_Float16)((1.0f / 256.0f) * __cosf((float)m * (PI_F / 512.0f)));
  }
}

// ---- channel transform: Y[b] = A(64x64) @ X[b] (64 x 65536), fp32 WMMA ----
// Per wave: one 16-wide N stripe, full M=64 (4 C-tiles), K=64 in 16 steps of 4.
// When Yh16 != null, also stores an f16 copy of Y (TDM staging source).
__global__ void chan_gemm(const float* __restrict__ Xin,
                          const float* __restrict__ Amat,
                          const float* __restrict__ gate,
                          float* __restrict__ Yout,
                          _Float16* __restrict__ Yh16, int use_gate) {
  int lane = threadIdx.x & 31;
  int wv   = threadIdx.x >> 5;
  int nt   = blockIdx.x * 8 + wv;                   // global n-tile id
  int b    = nt >> 12;                              // 4096 tiles per batch
  int n0   = (nt & 4095) << 4;
  int hl   = lane >> 4, col = lane & 15;

  const float* xb = Xin + (size_t)b * CHW + n0 + col;
  float g = 1.0f;
  if (use_gate) g = gate[(size_t)b * HW + n0 + col];

  v8f acc[4] = {};
  #pragma unroll
  for (int ks = 0; ks < 16; ++ks) {
    int c0 = ks * 4 + 2 * hl;                       // B rows: K = v + 2*half
    v2f bfrag;
    bfrag.x = xb[(size_t)(c0 + 0) * HW] * g;
    bfrag.y = xb[(size_t)(c0 + 1) * HW] * g;
    #pragma unroll
    for (int m = 0; m < 4; ++m) {                   // A rows: M = lane%16
      v2f afrag = *(const v2f*)(Amat + (m * 16 + col) * 64 + c0);
      acc[m] = __builtin_amdgcn_wmma_f32_16x16x4_f32(
          false, afrag, false, bfrag, (short)0, acc[m], false, false);
    }
  }
  size_t base = (size_t)b * CHW + n0 + col;         // C/D: row = r + 8*half
  #pragma unroll
  for (int m = 0; m < 4; ++m)
    #pragma unroll
    for (int r = 0; r < 8; ++r) {
      size_t idx = base + (size_t)(m * 16 + r + 8 * hl) * HW;
      Yout[idx] = acc[m][r];
      if (Yh16) Yh16[idx] = (_Float16)acc[m][r];
    }
}

// ---- per-pixel top-8 over 64 channels ------------------------------------
__global__ void topk_chan(const float* __restrict__ ca, float* __restrict__ ctk) {
  int tid = blockIdx.x * 256 + threadIdx.x;         // B_*HW threads
  int b = tid >> 16, n = tid & 65535;
  const float* p = ca + (size_t)b * CHW + n;
  float t[8];
  #pragma unroll
  for (int i = 0; i < 8; ++i) t[i] = -3.4e38f;
  #pragma unroll 4
  for (int c = 0; c < 64; ++c) top8_insert(t, p[(size_t)c * HW]);
  float* q = ctk + (size_t)b * 8 * HW + n;
  #pragma unroll
  for (int i = 0; i < 8; ++i) q[(size_t)i * HW] = t[i];
}

// ---- 3x3 conv (8->1, SAME) + sigmoid -> gate ------------------------------
__global__ void conv_gate(const float* __restrict__ ctk, const float* __restrict__ cw,
                          const float* __restrict__ cb, float* __restrict__ gate) {
  int tid = blockIdx.x * 256 + threadIdx.x;         // B_*HW
  int b = tid >> 16, hw = tid & 65535;
  int h = hw >> 8, w = hw & 255;
  float s = cb[0];
  #pragma unroll
  for (int i = 0; i < 8; ++i) {
    const float* plane = ctk + (size_t)(b * 8 + i) * HW;
    #pragma unroll
    for (int dy = 0; dy < 3; ++dy) {
      int hh = h + dy - 1;
      if ((unsigned)hh < 256u) {
        #pragma unroll
        for (int dx = 0; dx < 3; ++dx) {
          int ww = w + dx - 1;
          if ((unsigned)ww < 256u)
            s += plane[hh * 256 + ww] * cw[i * 9 + dy * 3 + dx];
        }
      }
    }
  }
  gate[tid] = 1.0f / (1.0f + __expf(-s));
}

// ---- fused 2D spatial DCT + per-(b,c) top-8, one WG per 256x256 image -----
// Image staged into LDS by the Tensor Data Mover (with HW row padding to the
// bank-skewed LDX pitch); both GEMMs read contiguous b128 WMMA fragments.
__global__ void spatial_dct_topk(const _Float16* __restrict__ xh,
                                 const _Float16* __restrict__ Drow,
                                 float* __restrict__ topc) {
  extern __shared__ _Float16 smem[];
  _Float16* Xh = smem;                              // 256*LDX halves (dyn-LDS base)
  _Float16* Tt = smem + 256 * LDX;                  // 256*LDX halves
  int bc = blockIdx.x;
  int t = threadIdx.x;
  int lane = t & 31, wv = t >> 5;
  int hl = lane >> 4, col = lane & 15;

  __builtin_prefetch(Drow + t * 64, 0, 1);          // warm 128KB DCT matrix in L2

  // Stage A: f16 image -> LDS (padded rows)
#if HAVE_TDM
  if (wv == 0) {
    unsigned long long ga = (unsigned long long)(const void*)(xh + (size_t)bc * HW);
    // D# group0: count=1 | lds_addr=0 | global_addr | type=2
    u32x4 g0 = { 1u, 0u, (unsigned)ga, (unsigned)(ga >> 32) | (2u << 30) };
    // D# group1: data_size=2B, pad_enable, pad_interval=128dw (code 6),
    // pad_amount=4dw (code 3); tensor 256x256, tile 256x256, dim0_stride=256
    i32x8 g1 = { 0x07910000, 0x01000000, 0x01000000, 0x01000000,
                 0x00000100, 0x00000100, 0, 0 };
    i32x4 gz = { 0, 0, 0, 0 };
#if TDM_6ARG
    i32x8 gz8 = { 0, 0, 0, 0, 0, 0, 0, 0 };
    __builtin_amdgcn_tensor_load_to_lds(g0, g1, gz, gz, gz8, 0);
#else
    __builtin_amdgcn_tensor_load_to_lds(g0, g1, gz, gz, 0);
#endif
  }
  __builtin_amdgcn_s_wait_tensorcnt(0);             // wave0 drains its TDM op
#else
  for (int i = 0; i < 256; ++i)                     // manual staging fallback
    Xh[i * LDX + t] = xh[(size_t)bc * HW + i * 256 + t];
#endif
  __syncthreads();

  // Stage B: T[h,k] = sum_n Xh[h,n]*Drow[k,n]; store transposed Tt[k,h]
  for (int tile = wv; tile < 256; tile += 8) {
    int m0 = (tile >> 4) << 4;                      // h block
    int k0 = (tile & 15) << 4;                      // freq block
    v8f acc = {};
    #pragma unroll
    for (int s = 0; s < 8; ++s) {
      int n0 = s * 32;
      V16 a, bb;
      const _Float16* pa = Xh + (m0 + col) * LDX + n0 + hl * 8;   // A: K chunks
      a.h[0] = *(const v8h*)pa;
      a.h[1] = *(const v8h*)(pa + 16);
      const _Float16* pb = Drow + (k0 + col) * 256 + n0 + hl * 8;  // B: col=lane%16
      bb.h[0] = *(const v8h*)pb;
      bb.h[1] = *(const v8h*)(pb + 16);
      acc = __builtin_amdgcn_wmma_f32_16x16x32_f16(
          false, a.v, false, bb.v, (short)0, acc, false, false);
    }
    v8h st;
    #pragma unroll
    for (int r = 0; r < 8; ++r) st[r] = (_Float16)acc[r];
    *(v8h*)(Tt + (k0 + col) * LDX + m0 + hl * 8) = st;  // transposed, contiguous
  }
  __syncthreads();

  // Stage C: dct2[k2,k] = sum_h Drow[k2,h]*Tt[k,h]; keep running top-8
  float top8[8];
  #pragma unroll
  for (int i = 0; i < 8; ++i) top8[i] = -3.4e38f;
  for (int tile = wv; tile < 256; tile += 8) {
    int m0 = (tile >> 4) << 4;                      // k2 block
    int k0 = (tile & 15) << 4;                      // k block
    v8f acc = {};
    #pragma unroll
    for (int s = 0; s < 8; ++s) {
      int n0 = s * 32;                              // h
      V16 a, bb;
      const _Float16* pa = Drow + (m0 + col) * 256 + n0 + hl * 8;
      a.h[0] = *(const v8h*)pa;
      a.h[1] = *(const v8h*)(pa + 16);
      const _Float16* pb = Tt + (k0 + col) * LDX + n0 + hl * 8;
      bb.h[0] = *(const v8h*)pb;
      bb.h[1] = *(const v8h*)(pb + 16);
      acc = __builtin_amdgcn_wmma_f32_16x16x32_f16(
          false, a.v, false, bb.v, (short)0, acc, false, false);
    }
    #pragma unroll
    for (int r = 0; r < 8; ++r) top8_insert(top8, acc[r]);
  }
  // Block reduction: 256 sorted 8-lists -> one top-8 (reuse Xh region)
  float* red = (float*)Xh;
  #pragma unroll
  for (int i = 0; i < 8; ++i) red[t * 8 + i] = top8[i];
  __syncthreads();
  if (t == 0) {
    float m[8];
    #pragma unroll
    for (int i = 0; i < 8; ++i) m[i] = -3.4e38f;
    for (int j = 0; j < 2048; ++j) top8_insert(m, red[j]);
    #pragma unroll
    for (int i = 0; i < 8; ++i) topc[bc * 8 + i] = m[i];   // descending, [B,C*8]
  }
}

// ---- tiny MLP: relu(topc@fc1^T) -> sigmoid(@fc2^T) -> per-(b,c) weight ----
__global__ void mlp_gate(const float* __restrict__ topc, const float* __restrict__ fc1,
                         const float* __restrict__ fc2, float* __restrict__ wvec) {
  __shared__ float hsh[32];
  int t = threadIdx.x;
  if (t < 32) {
    int b = t >> 3, j = t & 7;
    float s = 0.f;
    for (int m = 0; m < 512; ++m) s += topc[b * 512 + m] * fc1[j * 512 + m];
    hsh[t] = fmaxf(s, 0.f);
  }
  __syncthreads();
  int b = t >> 6, c = t & 63;
  float s = 0.f;
  #pragma unroll
  for (int j = 0; j < 8; ++j) s += hsh[b * 8 + j] * fc2[c * 8 + j];
  wvec[t] = 1.0f / (1.0f + __expf(-s));
}

// ---- out = x * w[b,c] (vectorized) ----------------------------------------
__global__ void scale_out(const float* __restrict__ x, const float* __restrict__ wvec,
                          float* __restrict__ out) {
  size_t i4 = ((size_t)blockIdx.x * 256 + threadIdx.x) * 4;
  float w = wvec[(int)(i4 >> 16)];                  // flat (b*64+c) index
  v4f v = *(const v4f*)(x + i4);
  v4f o = {v.x * w, v.y * w, v.z * w, v.w * w};
  *(v4f*)(out + i4) = o;
}

extern "C" void kernel_launch(void* const* d_in, const int* in_sizes, int n_in,
                              void* d_out, int out_size, void* d_ws, size_t ws_size,
                              hipStream_t stream) {
  (void)in_sizes; (void)n_in; (void)out_size; (void)ws_size;
  const float* ip  = (const float*)d_in[0];
  const float* fc1 = (const float*)d_in[1];
  const float* fc2 = (const float*)d_in[2];
  const float* cw  = (const float*)d_in[3];
  const float* cb  = (const float*)d_in[4];
  float* out = (float*)d_out;

  // ws layout (16B-aligned): ca | x | ctk | gate | Dc | Minv | Drow | topc | wvec | xh16
  float*    ca   = (float*)d_ws;
  float*    xbuf = ca + (size_t)TOT;
  float*    ctk  = xbuf + (size_t)TOT;
  float*    gate = ctk + (size_t)B_ * 8 * HW;
  float*    Dc   = gate + (size_t)B_ * HW;
  float*    Minv = Dc + 64 * 64;
  _Float16* Drow = (_Float16*)(Minv + 64 * 64);
  float*    topc = (float*)(Drow + 256 * 256);
  float*    wvec = topc + B_ * C_ * 8;
  _Float16* xh16 = (_Float16*)(wvec + B_ * C_);

  gen_mats<<<256, 256, 0, stream>>>(Dc, Minv, Drow);
  chan_gemm<<<2048, 256, 0, stream>>>(ip, Dc, nullptr, ca, nullptr, 0);   // ca = Dc@ip
  topk_chan<<<1024, 256, 0, stream>>>(ca, ctk);
  conv_gate<<<1024, 256, 0, stream>>>(ctk, cw, cb, gate);
  chan_gemm<<<2048, 256, 0, stream>>>(ca, Minv, gate, xbuf, xh16, 1);     // x = Minv@(ca*g)
  spatial_dct_topk<<<256, 256, 2 * 256 * LDX * sizeof(_Float16), stream>>>(xh16, Drow, topc);
  mlp_gate<<<1, 256, 0, stream>>>(topc, fc1, fc2, wvec);
  scale_out<<<TOT / 1024, 256, 0, stream>>>(xbuf, wvec, out);
}
